// TRM_14499809591638
// MI455X (gfx1250) — matrix operand order repaired and checked by reading, not compiled, not run on hardware
//
#include <hip/hip_runtime.h>
#include <cstddef>
#include <cstdint>

// ---------------------------------------------------------------------------
// TRM-style recurrent refinement for MI455X (gfx1250, wave32, WMMA + TDM).
//
// Dominant work after algebraic restructuring: 33 GEMMs of 8192x1024x1024,
// bf16 operands / f32 accumulate on v_wmma_f32_16x16x32_bf16. Weights (6 MB)
// and activations (16 MB bf16) are L2-resident (192 MB L2), so the loop is
// tensor-op bound. Data movement uses the Tensor Data Mover: per 128x128x64
// stage, wave 0 issues two tensor_load_to_lds DMAs (A tile, W tile) into a
// ping-pong LDS buffer and the whole block overlaps DMA with 16 WMMAs/wave.
// ---------------------------------------------------------------------------

typedef __bf16 bf16;
typedef __attribute__((ext_vector_type(16))) __bf16 v16bf;
typedef __attribute__((ext_vector_type(8)))  __bf16 v8bf;
typedef __attribute__((ext_vector_type(8)))  float  v8f;
typedef __attribute__((ext_vector_type(4)))  unsigned int u32x4;
typedef __attribute__((ext_vector_type(8)))  int i32x8;
typedef __attribute__((ext_vector_type(4)))  int i32x4;

union FragAB { v16bf f; v8bf h[2]; };

#if defined(__gfx1250__) && __has_builtin(__builtin_amdgcn_tensor_load_to_lds)
#define USE_TDM 1
#else
#define USE_TDM 0
#endif

#if USE_TDM
#if __has_include(<hip/amd_detail/amd_gfx1250_TDM.h>)
// amdgpu-toolchain (clang-23 / therock): 6-arg form
#define TDM_LOAD(g0, g1, z4, z8) __builtin_amdgcn_tensor_load_to_lds(g0, g1, z4, z4, z8, 0)
#else
// ROCm 7.2 (clang-22): 5-arg form
#define TDM_LOAD(g0, g1, z4, z8) __builtin_amdgcn_tensor_load_to_lds(g0, g1, z4, z4, 0)
#endif
#endif

#define BM   128
#define BN   128
#define BK   64
#define KSUB 32

// Issue one TDM 2D tile load: tile_h rows x tile_w bf16 elements, row stride
// `stride_elems`, from gptr into LDS byte offset lds_off. Uniform args only.
__device__ __forceinline__ void tdm_load_2d(unsigned lds_off, const bf16* gptr,
                                            unsigned tile_w, unsigned tile_h,
                                            unsigned stride_elems)
{
#if USE_TDM
    unsigned long long ga = (unsigned long long)(uintptr_t)gptr;
    u32x4 g0;
    g0[0] = 1u;                                       // count=1, user descriptor
    g0[1] = lds_off;                                  // D#.lds_addr (bytes)
    g0[2] = (unsigned)(ga & 0xffffffffu);             // global_addr[31:0]
    g0[3] = (unsigned)((ga >> 32) & 0x01ffffffu)      // global_addr[56:32]
            | (2u << 30);                             // type=2 ("image")
    i32x8 g1;
    g1[0] = (int)(1u << 16);                          // data_size=1 (2 bytes), mask=0
    g1[1] = (int)((tile_w & 0xffffu) << 16);          // tensor_dim0[15:0] (=tile_w: no OOB)
    g1[2] = (int)((tile_w >> 16) | ((tile_h & 0xffffu) << 16));  // dim0 hi | dim1 lo
    g1[3] = (int)((tile_h >> 16) | ((tile_w & 0xffffu) << 16));  // dim1 hi | tile_dim0
    g1[4] = (int)(tile_h & 0xffffu);                  // tile_dim1; tile_dim2=0
    g1[5] = (int)stride_elems;                        // tensor_dim0_stride[31:0]
    g1[6] = 0;                                        // stride hi / dim1_stride lo
    g1[7] = 0;
    i32x4 z4 = {};
    i32x8 z8 = {};
    (void)z8;
    TDM_LOAD(g0, g1, z4, z8);
#else
    (void)lds_off; (void)gptr; (void)tile_w; (void)tile_h; (void)stride_elems;
#endif
}

// C[M,N] = act( A[M,K](bf16) @ W[N,K](bf16)^T + bias[N] + add1[M,N] + add2[M,N] )
// act: 0=none, 1=tanh, 2=sigmoid. outF / outH nullable. M = gridDim.y * BM.
__global__ __launch_bounds__(256) void wmma_gemm_bf16(
    const bf16* __restrict__ A, int lda,
    const bf16* __restrict__ W, int ldw,
    const float* __restrict__ bias,
    const float* __restrict__ add1,
    const float* __restrict__ add2,
    float* __restrict__ outF,
    bf16*  __restrict__ outH,
    int N, int K, int act)
{
    __shared__ bf16 sA[2][BM * BK];   // 2 x 16 KB
    __shared__ bf16 sW[2][BN * BK];   // 2 x 16 KB

    const int tid  = threadIdx.x;
    const int lane = tid & 31;
    const int wave = tid >> 5;
    const int wm   = (wave & 3) * 32;    // wave M offset in block tile (4 waves)
    const int wn   = (wave >> 2) * 64;   // wave N offset in block tile (2 waves)
    const int m0   = blockIdx.y * BM;
    const int n0   = blockIdx.x * BN;

    v8f acc[2][4] = {};

    // WMMA fragment lane mapping (16-bit A 16x32 / B 32x16, wave32).
    const int mA  = lane & 15;
    const int kbA = (lane >> 4) * 8;     // A frag: K bases {kbA, kbA+16}
    const int nB  = lane & 15;
    const int kbB = (lane >> 4) * 8;     // B frag: K bases {kbB, kbB+16}

    auto compute_stage = [&](const bf16* cA, const bf16* cW) {
        #pragma unroll
        for (int ks = 0; ks < BK; ks += KSUB) {
            FragAB a[2], b[4];
            #pragma unroll
            for (int i = 0; i < 2; ++i) {
                const bf16* pa = &cA[(wm + i * 16 + mA) * BK + ks + kbA];
                a[i].h[0] = *(const v8bf*)(pa);
                a[i].h[1] = *(const v8bf*)(pa + 16);
            }
            #pragma unroll
            for (int j = 0; j < 4; ++j) {
                const bf16* pb = &cW[(wn + j * 16 + nB) * BK + ks + kbB];
                b[j].h[0] = *(const v8bf*)(pb);
                b[j].h[1] = *(const v8bf*)(pb + 16);
            }
            #pragma unroll
            for (int i = 0; i < 2; ++i)
                #pragma unroll
                for (int j = 0; j < 4; ++j)
                    acc[i][j] = __builtin_amdgcn_wmma_f32_16x16x32_bf16(
                        false, a[i].f, false, b[j].f, (short)0, acc[i][j], false, false);
        }
    };

    const int nstage = K / BK;

#if USE_TDM
    if (wave == 0) {
        tdm_load_2d((unsigned)(uintptr_t)&sA[0][0], &A[(size_t)m0 * lda], BK, BM, (unsigned)lda);
        tdm_load_2d((unsigned)(uintptr_t)&sW[0][0], &W[(size_t)n0 * ldw], BK, BN, (unsigned)ldw);
    }
    for (int s = 0; s < nstage; ++s) {
        const int cur = s & 1;
        if (wave == 0) {
            if (s + 1 < nstage) {
                const int k1 = (s + 1) * BK;
                tdm_load_2d((unsigned)(uintptr_t)&sA[cur ^ 1][0], &A[(size_t)m0 * lda + k1], BK, BM, (unsigned)lda);
                tdm_load_2d((unsigned)(uintptr_t)&sW[cur ^ 1][0], &W[(size_t)n0 * ldw + k1], BK, BN, (unsigned)ldw);
                __builtin_amdgcn_s_wait_tensorcnt(2);  // stage s's two DMAs done (in-order)
            } else {
                __builtin_amdgcn_s_wait_tensorcnt(0);
            }
        }
        __syncthreads();                 // stage s tiles visible to all waves
        compute_stage(sA[cur], sW[cur]);
        __syncthreads();                 // all waves done reading before overwrite
    }
#else
    // Fallback: synchronous cooperative global -> LDS staging.
    for (int s = 0; s < nstage; ++s) {
        const int cur = s & 1;
        const int k0 = s * BK;
        __syncthreads();
        #pragma unroll
        for (int p = 0; p < 4; ++p) {
            const int e = (p * 256 + tid) * 8;     // 8192 elems per tile, 16B/thread/pass
            const int r = e >> 6;
            const int c = e & 63;
            *(v8bf*)&sA[cur][r * BK + c] = *(const v8bf*)&A[(size_t)(m0 + r) * lda + k0 + c];
            *(v8bf*)&sW[cur][r * BK + c] = *(const v8bf*)&W[(size_t)(n0 + r) * ldw + k0 + c];
        }
        if (k0 + BK < K)
            __builtin_prefetch(&A[(size_t)(m0 + (tid >> 3)) * lda + k0 + BK], 0, 3);
        __syncthreads();
        compute_stage(sA[cur], sW[cur]);
    }
#endif

    // Epilogue. C/D 16x16 f32 layout: lanes 0-15 hold M=r, lanes 16-31 M=r+8.
    const int colBase = n0 + wn + (lane & 15);
    const int rowHalf = (lane >> 4) * 8;
    for (int i = 0; i < 2; ++i) {
        for (int j = 0; j < 4; ++j) {
            const int col = colBase + j * 16;
            const float bv = bias ? bias[col] : 0.0f;
            for (int r = 0; r < 8; ++r) {
                const int row = m0 + wm + i * 16 + rowHalf + r;
                const size_t idx = (size_t)row * N + col;
                float v = acc[i][j][r] + bv;
                if (add1) v += add1[idx];
                if (add2) v += add2[idx];
                if (act == 1)      v = tanhf(v);
                else if (act == 2) v = 1.0f / (1.0f + expf(-v));
                if (outF) outF[idx] = v;
                if (outH) outH[idx] = (bf16)v;
            }
        }
    }
}

__global__ void cvt_f32_to_bf16(const float* __restrict__ in, bf16* __restrict__ out, int n) {
    int i = blockIdx.x * blockDim.x + threadIdx.x;
    if (i < n) out[i] = (bf16)in[i];
}

extern "C" void kernel_launch(void* const* d_in, const int* in_sizes, int n_in,
                              void* d_out, int out_size, void* d_ws, size_t ws_size,
                              hipStream_t stream) {
    (void)in_sizes; (void)n_in; (void)out_size; (void)ws_size;

    constexpr int B  = 8192;
    constexpr int DX = 512;
    constexpr int DZ = 1024;
    constexpr int DY = 256;
    constexpr int T  = 4;   // python scalar in setup_inputs(); fixed for this problem
    constexpr int NI = 6;   // python scalar "n"

    const float* xs  = (const float*)d_in[0];
    const float* Wx  = (const float*)d_in[1];
    const float* bx  = (const float*)d_in[2];
    const float* Wy  = (const float*)d_in[3];
    const float* by  = (const float*)d_in[4];
    const float* Wn  = (const float*)d_in[5];
    const float* bn  = (const float*)d_in[6];
    const float* Wu  = (const float*)d_in[7];
    const float* bu  = (const float*)d_in[8];
    const float* Wq  = (const float*)d_in[9];
    const float* bq  = (const float*)d_in[10];
    const float* ysi = (const float*)d_in[11];
    const float* zsi = (const float*)d_in[12];

    // ---- workspace carve-up (256B aligned) ----
    char* ws = (char*)d_ws;
    size_t off = 0;
    auto allocB = [&](size_t elems) -> bf16* {
        off = (off + 255) & ~(size_t)255;
        bf16* p = (bf16*)(ws + off);
        off += elems * sizeof(bf16);
        return p;
    };
    auto allocF = [&](size_t elems) -> float* {
        off = (off + 255) & ~(size_t)255;
        float* p = (float*)(ws + off);
        off += elems * sizeof(float);
        return p;
    };

    bf16* Wx_h  = allocB((size_t)DZ * DX);
    bf16* Wy_h  = allocB((size_t)DZ * DY);
    bf16* Wn_h  = allocB((size_t)DZ * 3 * DZ);
    bf16* Wu_h  = allocB((size_t)DY * DZ);
    bf16* Wq_h  = allocB((size_t)DY * DZ);
    bf16* xs_h  = allocB((size_t)B * DX);
    bf16* ysi_h = allocB((size_t)B * DY);
    bf16* xse_h = allocB((size_t)B * DZ);   // xs_e (constant)
    bf16* ys_h  = allocB((size_t)B * DZ);   // current ys_e
    bf16* zsA_h = allocB((size_t)B * DZ);   // z ping
    bf16* zsB_h = allocB((size_t)B * DZ);   // z pong
    float* cx   = allocF((size_t)B * DZ);   // xs_e @ Wnx^T
    float* cyb  = allocF((size_t)B * DZ);   // ys_e @ Wny^T + bn

    auto cvt = [&](const float* in, bf16* out, size_t n) {
        cvt_f32_to_bf16<<<dim3((unsigned)((n + 255) / 256)), dim3(256), 0, stream>>>(in, out, (int)n);
    };
    auto gemm = [&](const bf16* A, int lda, const bf16* W, int ldw,
                    const float* bias, const float* add1, const float* add2,
                    float* outF, bf16* outH, int N, int K, int act) {
        dim3 grid((unsigned)(N / BN), (unsigned)(B / BM));
        wmma_gemm_bf16<<<grid, dim3(256), 0, stream>>>(A, lda, W, ldw, bias, add1, add2,
                                                       outF, outH, N, K, act);
    };

    // ---- convert all fp32 operands to bf16 once ----
    cvt(Wx,  Wx_h,  (size_t)DZ * DX);
    cvt(Wy,  Wy_h,  (size_t)DZ * DY);
    cvt(Wn,  Wn_h,  (size_t)DZ * 3 * DZ);
    cvt(Wu,  Wu_h,  (size_t)DY * DZ);
    cvt(Wq,  Wq_h,  (size_t)DY * DZ);
    cvt(xs,  xs_h,  (size_t)B * DX);
    cvt(ysi, ysi_h, (size_t)B * DY);
    cvt(zsi, zsA_h, (size_t)B * DZ);

    // ---- encoders ----
    gemm(xs_h,  DX, Wx_h, DX, bx, nullptr, nullptr, nullptr, xse_h, DZ, DX, 0);  // xs_e
    gemm(ysi_h, DY, Wy_h, DY, by, nullptr, nullptr, nullptr, ys_h,  DZ, DY, 0);  // ys_e
    gemm(xse_h, DZ, Wn_h + 0 * DZ, 3 * DZ, nullptr, nullptr, nullptr, cx, nullptr, DZ, DZ, 0); // cx

    // ---- T outer iterations: 1 y-part GEMM + NI z-steps + 1 y-step ----
    bf16* zc = zsA_h;
    bf16* za = zsB_h;
    for (int t = 0; t < T; ++t) {
        gemm(ys_h, DZ, Wn_h + 1 * DZ, 3 * DZ, bn, nullptr, nullptr, cyb, nullptr, DZ, DZ, 0);
        for (int i = 0; i < NI; ++i) {
            gemm(zc, DZ, Wn_h + 2 * DZ, 3 * DZ, nullptr, cx, cyb, nullptr, za, DZ, DZ, 1);
            bf16* tmp = zc; zc = za; za = tmp;
        }
        gemm(zc, DZ, Wn_h + 2 * DZ, 3 * DZ, nullptr, cyb, nullptr, nullptr, ys_h, DZ, DZ, 1);
    }

    // ---- heads ----
    float* y_hat = (float*)d_out;
    float* q_hat = y_hat + (size_t)B * DY;
    gemm(ys_h, DZ, Wu_h, DZ, bu, nullptr, nullptr, y_hat, nullptr, DY, DZ, 2);
    gemm(ys_h, DZ, Wq_h, DZ, bq, nullptr, nullptr, q_hat, nullptr, DY, DZ, 2);
}
